// BahdanauAttention_68204080661120
// MI455X (gfx1250) — matrix operand order repaired and checked by reading, not compiled, not run on hardware
//
#include <hip/hip_runtime.h>
#include <hip/hip_bf16.h>

#define BATCH 32
#define SEQ   2048
#define EMB   1024
#define ADIM  1024
#define ROWS_PER_BLK 64
#define LDS_STRIDE   1032   // ushorts per row: 1024 + 8 pad (16B aligned, rotates banks)

typedef __attribute__((ext_vector_type(16))) __bf16        v16bf;
typedef __attribute__((ext_vector_type(8)))  float         v8f;
typedef __attribute__((ext_vector_type(4)))  unsigned int  u32x4;

union Frag { u32x4 u[2]; v16bf v; };

__device__ __forceinline__ unsigned int pack_bf16x2(float lo, float hi) {
    unsigned int ulo = __float_as_uint(lo);
    unsigned int uhi = __float_as_uint(hi);
    unsigned int rlo = (ulo + 0x7FFFu + ((ulo >> 16) & 1u)) >> 16;
    unsigned int rhi = (uhi + 0x7FFFu + ((uhi >> 16) & 1u)) & 0xFFFF0000u;
    return rhi | rlo;
}

__device__ __forceinline__ float fast_tanh(float x) {
    float t;
    asm volatile("v_tanh_f32 %0, %1" : "=v"(t) : "v"(x));   // gfx1250 transcendental
    return t;
}

// --- Kernel 1: W_enc fp32 -> bf16 (L2-resident operand for WMMA B fragments) ---
__global__ __launch_bounds__(256) void cvt_bf16_kernel(const float* __restrict__ in,
                                                       unsigned short* __restrict__ out,
                                                       int n) {
    int i = (blockIdx.x * 256 + threadIdx.x) * 2;
    if (i < n) {
        float2 f = *reinterpret_cast<const float2*>(in + i);
        *reinterpret_cast<unsigned int*>(out + i) = pack_bf16x2(f.x, f.y);
    }
}

// --- Kernel 2: pre[b,a] = dot(h[b,:], W_dec[a,:]) + b_dec[a] + b_enc[a] (tiny GEMV) ---
__global__ __launch_bounds__(256) void decpre_kernel(const float* __restrict__ h,
                                                     const float* __restrict__ Wd,
                                                     const float* __restrict__ bd,
                                                     const float* __restrict__ be,
                                                     float* __restrict__ pre) {
    int t = blockIdx.x * 256 + threadIdx.x;     // [0, 32*1024)
    int b = t >> 10, a = t & 1023;
    const float* hr = h  + b * EMB;
    const float* wr = Wd + a * EMB;
    float s = 0.f;
    #pragma unroll 4
    for (int d = 0; d < EMB; d += 4) {
        float4 hv = *reinterpret_cast<const float4*>(hr + d);
        float4 wv = *reinterpret_cast<const float4*>(wr + d);
        s += hv.x * wv.x + hv.y * wv.y + hv.z * wv.z + hv.w * wv.w;
    }
    pre[t] = s + bd[a] + be[a];
}

// --- Kernel 3: fused scores = tanh(enc @ W_enc^T + pre) @ W_v via WMMA bf16 ---
// Block = 128 threads = 4 waves over 64 rows. 2x2 register blocking: each wave
// holds 2 A-frags x 2 B-frags -> 4 WMMAs per K-step (halves LDS/VMEM per WMMA).
// Waves {0,1} own rows 0-31, {2,3} rows 32-63; pair members take alternating
// 32-column N-groups and combine score partials through LDS ds_add_f32.
__global__ __launch_bounds__(128) void scores_wmma_kernel(const float* __restrict__ enc,
                                                          const unsigned short* __restrict__ Wb,
                                                          const float* __restrict__ pre,
                                                          const float* __restrict__ Wv,
                                                          float* __restrict__ scores) {
    extern __shared__ unsigned char smem_raw[];
    unsigned short* smem = (unsigned short*)smem_raw;                       // 64 x 1032 bf16
    float* sscore = (float*)(smem_raw + ROWS_PER_BLK * LDS_STRIDE * 2);     // 64 floats

    const int tid  = threadIdx.x;
    const long long row0 = (long long)blockIdx.x * ROWS_PER_BLK;
    const int b = (int)(row0 / SEQ);

    // Stage 64 encoder rows fp32 -> bf16 into LDS (coalesced float4 reads).
    const float* src = enc + row0 * EMB;
    for (int i = tid; i < ROWS_PER_BLK * EMB / 4; i += 128) {
        int idx = i * 4;
        int r = idx >> 10, c = idx & 1023;
        float4 f = *reinterpret_cast<const float4*>(src + idx);
        unsigned int p0 = pack_bf16x2(f.x, f.y);
        unsigned int p1 = pack_bf16x2(f.z, f.w);
        *reinterpret_cast<uint2*>(&smem[r * LDS_STRIDE + c]) = make_uint2(p0, p1);
    }
    if (tid < ROWS_PER_BLK) sscore[tid] = 0.f;
    __syncthreads();

    const int wave   = tid >> 5;
    const int lane   = tid & 31;
    const int laneLo = lane & 15;
    const int hi     = lane >> 4;
    const int mHalf  = wave >> 1;   // row group: mHalf*32 .. +31
    const int nPar   = wave & 1;    // N-group parity within the wave pair

    // A fragment bases (dense 16-bit A layout: per lane two 16B runs at +hi*8, +16).
    const unsigned short* aBase0 = smem + (mHalf * 32 + laneLo) * LDS_STRIDE + hi * 8;
    const unsigned short* aBase1 = aBase0 + 16 * LDS_STRIDE;

    float sp[16];
    #pragma unroll
    for (int r = 0; r < 16; ++r) sp[r] = 0.f;

    for (int nt = nPar * 2; nt < ADIM / 16; nt += 4) {   // handle 16-col tiles nt, nt+1
        const int n0 = nt * 16 + laneLo;
        const unsigned short* bBase0 = Wb + n0 * EMB + hi * 8;   // lane col = N, K runs at +hi*8, +16
        const unsigned short* bBase1 = bBase0 + 16 * EMB;
        if (nt + 4 < ADIM / 16) {                                // prefetch next N-group
            __builtin_prefetch(bBase0 + 64 * EMB, 0, 0);
            __builtin_prefetch(bBase1 + 64 * EMB, 0, 0);
        }

        v8f acc00 = {}, acc01 = {}, acc10 = {}, acc11 = {};
        #pragma unroll 2
        for (int kc = 0; kc < EMB / 32; ++kc) {
            Frag a0, a1, b0, b1;
            a0.u[0] = *reinterpret_cast<const u32x4*>(aBase0 + kc * 32);
            a0.u[1] = *reinterpret_cast<const u32x4*>(aBase0 + kc * 32 + 16);
            a1.u[0] = *reinterpret_cast<const u32x4*>(aBase1 + kc * 32);
            a1.u[1] = *reinterpret_cast<const u32x4*>(aBase1 + kc * 32 + 16);
            b0.u[0] = *reinterpret_cast<const u32x4*>(bBase0 + kc * 32);
            b0.u[1] = *reinterpret_cast<const u32x4*>(bBase0 + kc * 32 + 16);
            b1.u[0] = *reinterpret_cast<const u32x4*>(bBase1 + kc * 32);
            b1.u[1] = *reinterpret_cast<const u32x4*>(bBase1 + kc * 32 + 16);
            acc00 = __builtin_amdgcn_wmma_f32_16x16x32_bf16(false, a0.v, false, b0.v,
                                                            (short)0, acc00, false, false);
            acc01 = __builtin_amdgcn_wmma_f32_16x16x32_bf16(false, a0.v, false, b1.v,
                                                            (short)0, acc01, false, false);
            acc10 = __builtin_amdgcn_wmma_f32_16x16x32_bf16(false, a1.v, false, b0.v,
                                                            (short)0, acc10, false, false);
            acc11 = __builtin_amdgcn_wmma_f32_16x16x32_bf16(false, a1.v, false, b1.v,
                                                            (short)0, acc11, false, false);
        }

        // Fuse: score[m] += tanh(acc + pre[b,n]) * Wv[n]; b_v dropped (softmax-invariant).
        const float pre0 = pre[b * ADIM + n0];
        const float pre1 = pre[b * ADIM + n0 + 16];
        const float wv0  = Wv[n0];
        const float wv1  = Wv[n0 + 16];
        #pragma unroll
        for (int r = 0; r < 8; ++r) {
            sp[r]     += fast_tanh(acc00[r] + pre0) * wv0;
            sp[r]     += fast_tanh(acc01[r] + pre1) * wv1;
            sp[r + 8] += fast_tanh(acc10[r] + pre0) * wv0;
            sp[r + 8] += fast_tanh(acc11[r] + pre1) * wv1;
        }
    }

    // Reduce the 16 N-lanes of each half-wave (wave32 shuffles).
    #pragma unroll
    for (int r = 0; r < 16; ++r) {
        float v = sp[r];
        v += __shfl_xor(v, 1, 32);
        v += __shfl_xor(v, 2, 32);
        v += __shfl_xor(v, 4, 32);
        v += __shfl_xor(v, 8, 32);
        sp[r] = v;
    }
    if (laneLo == 0) {
        #pragma unroll
        for (int r = 0; r < 8; ++r) {
            // C/D layout: lanes 16-31 hold M = r+8
            atomicAdd(&sscore[mHalf * 32 + r + hi * 8],      sp[r]);       // ds_add_f32
            atomicAdd(&sscore[mHalf * 32 + 16 + r + hi * 8], sp[r + 8]);
        }
    }
    __syncthreads();
    if (tid < ROWS_PER_BLK) scores[row0 + tid] = sscore[tid];
}

// --- Kernel 4: softmax over S per batch ---
__global__ __launch_bounds__(256) void softmax_kernel(const float* __restrict__ scores,
                                                      float* __restrict__ alpha) {
    __shared__ float red[256];
    const int b = blockIdx.x, tid = threadIdx.x;
    const float* s = scores + b * SEQ;
    float v[8], mx = -3.0e38f;
    #pragma unroll
    for (int i = 0; i < 8; ++i) { v[i] = s[tid + i * 256]; mx = fmaxf(mx, v[i]); }
    red[tid] = mx; __syncthreads();
    for (int off = 128; off > 0; off >>= 1) {
        if (tid < off) red[tid] = fmaxf(red[tid], red[tid + off]);
        __syncthreads();
    }
    mx = red[0]; __syncthreads();
    float sum = 0.f;
    #pragma unroll
    for (int i = 0; i < 8; ++i) { v[i] = __expf(v[i] - mx); sum += v[i]; }
    red[tid] = sum; __syncthreads();
    for (int off = 128; off > 0; off >>= 1) {
        if (tid < off) red[tid] += red[tid + off];
        __syncthreads();
    }
    float inv = 1.0f / red[0];
    float* a = alpha + b * SEQ;
    #pragma unroll
    for (int i = 0; i < 8; ++i) a[tid + i * 256] = v[i] * inv;
}

// --- Kernel 5: context[b,e] = sum_s alpha[b,s] * enc[b,s,e] (coalesced across e) ---
__global__ __launch_bounds__(256) void context_kernel(const float* __restrict__ enc,
                                                      const float* __restrict__ alpha,
                                                      float* __restrict__ ctx) {
    const int b = blockIdx.y;
    const int e = blockIdx.x * 256 + threadIdx.x;
    const float* ep = enc + (long long)b * SEQ * EMB + e;
    const float* ap = alpha + b * SEQ;
    float s = 0.f;
    #pragma unroll 4
    for (int t = 0; t < SEQ; ++t)
        s += ap[t] * ep[(long long)t * EMB];
    ctx[b * EMB + e] = s;
}

extern "C" void kernel_launch(void* const* d_in, const int* in_sizes, int n_in,
                              void* d_out, int out_size, void* d_ws, size_t ws_size,
                              hipStream_t stream) {
    const float* enc   = (const float*)d_in[0];   // [B,S,E]
    const float* dh    = (const float*)d_in[1];   // [B,D]
    const float* W_enc = (const float*)d_in[2];   // [A,E]
    const float* b_enc = (const float*)d_in[3];   // [A]
    const float* W_dec = (const float*)d_in[4];   // [A,D]
    const float* b_dec = (const float*)d_in[5];   // [A]
    const float* W_v   = (const float*)d_in[6];   // [1,A] flat
    (void)d_in[7];  // b_v: uniform shift, cancels in softmax; context/alpha unaffected

    // Workspace layout
    unsigned short* Wb = (unsigned short*)d_ws;                           // 2 MB bf16 W_enc
    float* pre    = (float*)((char*)d_ws + (size_t)ADIM * EMB * 2);       // 128 KB
    float* scores = pre + BATCH * ADIM;                                   // 256 KB
    float* ctx    = (float*)d_out;                                        // [B,E]
    float* alpha  = ctx + BATCH * EMB;                                    // [B,S]

    const size_t lds_bytes = (size_t)ROWS_PER_BLK * LDS_STRIDE * 2 + ROWS_PER_BLK * 4;

    cvt_bf16_kernel<<<(ADIM * EMB / 2 + 255) / 256, 256, 0, stream>>>(W_enc, Wb, ADIM * EMB);
    decpre_kernel<<<(BATCH * ADIM) / 256, 256, 0, stream>>>(dh, W_dec, b_dec, b_enc, pre);
    scores_wmma_kernel<<<(BATCH * SEQ) / ROWS_PER_BLK, 128, lds_bytes, stream>>>(
        enc, Wb, pre, W_v, scores);
    softmax_kernel<<<BATCH, 256, 0, stream>>>(scores, alpha);
    context_kernel<<<dim3(EMB / 256, BATCH), 256, 0, stream>>>(enc, alpha, ctx);
}
